// Multihead_Attention_28776280883372
// MI455X (gfx1250) — compile-verified, matrix operand order checked by reading, not run on hardware
//
#include <hip/hip_runtime.h>
#include <hip/hip_bf16.h>
#include <cstdint>

// Problem constants (from reference): B=4, S=2048, D=1024, H=16, hd=64
#define B_  4
#define S_  2048
#define D_  1024
#define H_  16
#define HD_ 64

typedef __attribute__((ext_vector_type(16))) __bf16 v16bf;
typedef __attribute__((ext_vector_type(8)))  float  v8f;
typedef __attribute__((ext_vector_type(4)))  unsigned int v4u;
typedef __attribute__((ext_vector_type(4)))  int  v4i;
typedef __attribute__((ext_vector_type(8)))  int  v8i;
typedef unsigned int u32;

// ---------------- helpers ----------------

__device__ __forceinline__ unsigned short f2bf(float f) {
  unsigned int u = __float_as_uint(f);
  u += 0x7fffu + ((u >> 16) & 1u);   // round-to-nearest-even
  return (unsigned short)(u >> 16);
}
__device__ __forceinline__ u32 pk2bf(float lo, float hi) {
  return (u32)f2bf(lo) | ((u32)f2bf(hi) << 16);
}
__device__ __forceinline__ float bf2f(unsigned short s) {
  return __uint_as_float(((unsigned int)s) << 16);
}

union FragU { v16bf v; uint4 q[2]; };

// Load one 16-lane x K32 bf16 fragment. Per ISA layout: lane = (row/col)&15,
// khalf = lane>>4 selects K in {khalf*8..+7} (elems 0-7) and {16+khalf*8..+7}
// (elems 8-15). rowptr points at element [row][k_chunk_base].
__device__ __forceinline__ v16bf ld_frag(const unsigned short* rowptr, int khalf) {
  FragU u;
  u.q[0] = *(const uint4*)(rowptr + khalf * 8);
  u.q[1] = *(const uint4*)(rowptr + 16 + khalf * 8);
  return u.v;
}

#define WMMA_BF16(A, Bm, C) \
  __builtin_amdgcn_wmma_f32_16x16x32_bf16(false, (A), false, (Bm), (short)0, (C), false, false)

// ---------------- Tensor Data Mover (async tile load) ----------------
// Contiguous bf16 blob -> LDS via TDM, descriptor per CDNA5 ISA:
// degenerate 2D tensor: one row of `nelems` 2-byte elements.
#if defined(__AMDGCN__) && __has_builtin(__builtin_amdgcn_tensor_load_to_lds) && \
    __has_builtin(__builtin_amdgcn_s_wait_tensorcnt)
#define HAVE_TDM 1
#endif

#ifdef HAVE_TDM
__device__ __forceinline__ void tdm_load_contig_bf16(u32 lds_off,
                                                     const unsigned short* src,
                                                     u32 nelems) {
  unsigned long long ga = (unsigned long long)(uintptr_t)src;
  // Group 0: count=1 (valid user D#), lds_addr, global_addr[56:0], type=2
  v4u g0 = { 1u,
             lds_off,
             (u32)ga,
             (u32)((ga >> 32) & 0x01ffffffu) | (2u << 30) };
  // Group 1: wg_mask=0, data_size=1 (2B); tensor_dim0=nelems, tensor_dim1=1;
  // tile_dim0=nelems, tile_dim1=1, tile_dim2=0; dim0_stride=nelems.
  v8i g1 = { (int)(1u << 16),
             (int)((nelems & 0xffffu) << 16),            // tensor_dim0[15:0]
             (int)((nelems >> 16) | (1u << 16)),         // dim0[31:16] | tensor_dim1 lo=1
             (int)((nelems & 0xffffu) << 16),            // dim1 hi=0 | tile_dim0
             (int)1,                                     // tile_dim1=1, tile_dim2=0
             (int)nelems,                                // tensor_dim0_stride lo32
             0, 0 };
  v4i gz = { 0, 0, 0, 0 };
#if __clang_major__ >= 23
  v8i gz8 = { 0, 0, 0, 0, 0, 0, 0, 0 };
  __builtin_amdgcn_tensor_load_to_lds(g0, g1, gz, gz, gz8, 0);
#else
  __builtin_amdgcn_tensor_load_to_lds(g0, g1, gz, gz, 0);
#endif
}
#endif

// One 64x64 bf16 tile (contiguous 8KB) global -> LDS.
// TDM path: wave 0 issues the DMA and waits TENSORcnt; caller barrier releases.
// Fallback: coalesced uint4 copy by the whole workgroup.
__device__ __forceinline__ void load_tile_bf16(unsigned short* dstLDS,
                                               const unsigned short* src,
                                               int tid) {
#ifdef HAVE_TDM
  if ((tid >> 5) == 0) {
    tdm_load_contig_bf16((u32)(uintptr_t)(void*)dstLDS, src, 64 * 64);
    __builtin_amdgcn_s_wait_tensorcnt(0);
  }
#else
  const uint4* s = (const uint4*)src;
  uint4* d = (uint4*)dstLDS;
  for (int i = tid; i < 512; i += 256) d[i] = s[i];
#endif
}

// ---------------- kernel 0: zero the amean region ----------------

__global__ __launch_bounds__(256) void zero_f32(float* __restrict__ p, int n4) {
  float4 z = {0.f, 0.f, 0.f, 0.f};
  for (int i = blockIdx.x * blockDim.x + threadIdx.x; i < n4;
       i += gridDim.x * blockDim.x)
    ((float4*)p)[i] = z;
}

// ---------------- kernel 1/2: QKV projection GEMM ----------------
// C[M,N] = X[M,K] @ W[K,N] + bias, written as bf16 into [B,H,S,hd] layout.
// mode 0: N=1024 -> q.   mode 1: N=2048 -> split into k (r<64) / v (r>=64).
// Block: 256 threads = 8 waves. Tile 64x64, BK=32. Wave w owns rows
// (w&3)*16..+15 and cols (w>>2)*32..+31 (two 16x16 WMMA accumulators).

__global__ __launch_bounds__(256) void qkv_gemm(
    const float* __restrict__ X, const float* __restrict__ W,
    const float* __restrict__ bias,
    unsigned short* __restrict__ out0, unsigned short* __restrict__ out1,
    int N, int mode)
{
  __shared__ __align__(16) unsigned short As[64 * 32];   // [row][k]
  __shared__ __align__(16) unsigned short Bs[64 * 32];   // [col][k] (W transposed)

  const int tid   = threadIdx.x;
  const int lane  = tid & 31;
  const int wave  = tid >> 5;
  const int m0    = (wave & 3) * 16;
  const int n0w   = (wave >> 2) * 32;
  const int rowBase = blockIdx.y * 64;
  const int colBase = blockIdx.x * 64;

  const int khalf = lane >> 4;
  const int ln    = lane & 15;

  v8f acc0 = {}; v8f acc1 = {};

  // staging assignments
  const int arow = tid >> 2;            // 0..63
  const int acol = (tid & 3) * 8;       // 0,8,16,24
  const int bk   = tid >> 3;            // 0..31
  const int bn0  = (tid & 7) * 8;       // 0..56

  for (int k0 = 0; k0 < D_; k0 += 32) {
    __syncthreads();
    // stage X tile (fp32 -> bf16, packed dword LDS stores)
    {
      const float* ap = X + (size_t)(rowBase + arow) * D_ + k0 + acol;
      float4 a0 = *(const float4*)ap;
      float4 a1 = *(const float4*)(ap + 4);
      u32* as = (u32*)(As + arow * 32 + acol);
      as[0] = pk2bf(a0.x, a0.y);
      as[1] = pk2bf(a0.z, a0.w);
      as[2] = pk2bf(a1.x, a1.y);
      as[3] = pk2bf(a1.z, a1.w);
    }
    // stage W tile transposed (fp32 -> bf16; transposed scatter)
    {
      const float* bp = W + (size_t)(k0 + bk) * N + colBase + bn0;
      float4 b0 = *(const float4*)bp;
      float4 b1 = *(const float4*)(bp + 4);
      Bs[(bn0 + 0) * 32 + bk] = f2bf(b0.x);
      Bs[(bn0 + 1) * 32 + bk] = f2bf(b0.y);
      Bs[(bn0 + 2) * 32 + bk] = f2bf(b0.z);
      Bs[(bn0 + 3) * 32 + bk] = f2bf(b0.w);
      Bs[(bn0 + 4) * 32 + bk] = f2bf(b1.x);
      Bs[(bn0 + 5) * 32 + bk] = f2bf(b1.y);
      Bs[(bn0 + 6) * 32 + bk] = f2bf(b1.z);
      Bs[(bn0 + 7) * 32 + bk] = f2bf(b1.w);
    }
    __syncthreads();

    v16bf af  = ld_frag(As + (m0 + ln) * 32, khalf);
    v16bf bf0 = ld_frag(Bs + (n0w + ln) * 32, khalf);
    v16bf bf1 = ld_frag(Bs + (n0w + 16 + ln) * 32, khalf);
    acc0 = WMMA_BF16(af, bf0, acc0);
    acc1 = WMMA_BF16(af, bf1, acc1);
  }

  // epilogue: +bias, bf16, remap into [B,H,S,hd]
  const int lm = khalf * 8;
#pragma unroll
  for (int t = 0; t < 2; ++t) {
    v8f acc = t ? acc1 : acc0;
    const int n = colBase + n0w + t * 16 + ln;
    const float bv = bias[n];
#pragma unroll
    for (int i = 0; i < 8; ++i) {
      const int gm = rowBase + m0 + lm + i;
      const int b  = gm >> 11;            // /S_
      const int s  = gm & (S_ - 1);
      const unsigned short val = f2bf(acc[i] + bv);
      if (mode == 0) {
        const int h = n >> 6, d = n & 63;
        out0[(((size_t)(b * H_ + h) * S_ + s) << 6) + d] = val;
      } else {
        const int h = n >> 7, r = n & 127;
        unsigned short* dst = (r < 64) ? out0 : out1;
        dst[(((size_t)(b * H_ + h) * S_ + s) << 6) + (r & 63)] = val;
      }
    }
  }
}

// ---------------- kernel 3: causal attention ----------------
// One workgroup per (b, 64-query-row tile). Loops over ALL 16 heads so the
// head-mean matrix stripe amean[b, r0:r0+64, :] is owned exclusively (plain
// global read-modify-write, no atomics).
// Pass 1: per-head per-row softmax max & 1/sum (stats in LDS).
// Pass 2: recompute scores, p = exp(s-m)/l; o += P@V (WMMA); amean += p/H.

__global__ __launch_bounds__(256) void attn(
    const unsigned short* __restrict__ Q,
    const unsigned short* __restrict__ K,
    const unsigned short* __restrict__ V,
    float* __restrict__ O, float* __restrict__ Amean)
{
  __shared__ __align__(16) unsigned short Qs[64 * 64];  // 8KB  [row][d]
  __shared__ __align__(16) unsigned short Ks[64 * 64];  // 8KB  [key][d]
  __shared__ __align__(16) unsigned short Vt[64 * 64];  // 8KB  [d][key]
  __shared__ __align__(16) float          Ss[64 * 64];  // 16KB scores
  __shared__ __align__(16) unsigned short Ps[64 * 64];  // 8KB  probs bf16
  __shared__ float Mst[H_ * 64];                        // 4KB row max
  __shared__ float Lst[H_ * 64];                        // 4KB 1/row sum

  const int tid  = threadIdx.x;
  const int lane = tid & 31;
  const int wave = tid >> 5;
  const int b    = blockIdx.y;
  const int tr   = blockIdx.x;
  const int r0   = tr * 64;
  const float scale = 0.125f;          // 1/sqrt(64)

  const int m0    = (wave & 3) * 16;
  const int nw    = (wave >> 2) * 32;
  const int khalf = lane >> 4;
  const int lm    = khalf * 8;
  const int ln    = lane & 15;

  // ================= Pass 1: softmax statistics =================
  for (int h = 0; h < H_; ++h) {
    const size_t bh = (size_t)(b * H_ + h) * S_;
    __syncthreads();                               // Qs free
    load_tile_bf16(Qs, Q + (bh + r0) * 64, tid);
    float rm = -__builtin_inff(), rl = 0.f;

    for (int j = 0; j <= tr; ++j) {
      __syncthreads();                             // Ks/Ss consumers done
      load_tile_bf16(Ks, K + (bh + (size_t)j * 64) * 64, tid);
      if (j < tr)
        __builtin_prefetch(K + (bh + (size_t)(j + 1) * 64) * 64 + tid * 16, 0, 1);
      __syncthreads();

      v8f s0 = {}; v8f s1 = {};
#pragma unroll
      for (int kk = 0; kk < 2; ++kk) {
        v16bf a  = ld_frag(Qs + (m0 + ln) * 64 + kk * 32, khalf);
        v16bf b0 = ld_frag(Ks + (nw + ln) * 64 + kk * 32, khalf);
        v16bf b1 = ld_frag(Ks + (nw + 16 + ln) * 64 + kk * 32, khalf);
        s0 = WMMA_BF16(a, b0, s0);
        s1 = WMMA_BF16(a, b1, s1);
      }
#pragma unroll
      for (int i = 0; i < 8; ++i) {
        Ss[(m0 + lm + i) * 64 + nw + ln]      = s0[i];
        Ss[(m0 + lm + i) * 64 + nw + 16 + ln] = s1[i];
      }
      __syncthreads();

      if (tid < 64) {                              // row owners
        const int gr = r0 + tid;
        const int nv = min(64, gr - j * 64 + 1);   // valid cols (causal)
        const float* srow = Ss + tid * 64;
        float tmax = -__builtin_inff();
        for (int n = 0; n < nv; ++n) tmax = fmaxf(tmax, srow[n] * scale);
        if (tmax > rm) { rl *= __expf(rm - tmax); rm = tmax; }
        float ls = 0.f;
        for (int n = 0; n < nv; ++n) ls += __expf(srow[n] * scale - rm);
        rl += ls;
      }
    }
    if (tid < 64) { Mst[h * 64 + tid] = rm; Lst[h * 64 + tid] = 1.0f / rl; }
  }

  // ================= Pass 2: output + head-mean =================
  for (int h = 0; h < H_; ++h) {
    const size_t bh = (size_t)(b * H_ + h) * S_;
    __syncthreads();                               // Qs free, stats visible
    load_tile_bf16(Qs, Q + (bh + r0) * 64, tid);
    v8f o0 = {}; v8f o1 = {};

    for (int j = 0; j <= tr; ++j) {
      __syncthreads();                             // Ks/Vt/Ps consumers done
      load_tile_bf16(Ks, K + (bh + (size_t)j * 64) * 64, tid);
      {
        const unsigned short* vsrc = V + (bh + (size_t)j * 64) * 64;
        for (int idx = tid; idx < 4096; idx += 256) {
          const int n = idx >> 6, d = idx & 63;
          Vt[d * 64 + n] = vsrc[idx];              // transpose into LDS
        }
      }
      __syncthreads();

      // scores
      v8f s0 = {}; v8f s1 = {};
#pragma unroll
      for (int kk = 0; kk < 2; ++kk) {
        v16bf a  = ld_frag(Qs + (m0 + ln) * 64 + kk * 32, khalf);
        v16bf b0 = ld_frag(Ks + (nw + ln) * 64 + kk * 32, khalf);
        v16bf b1 = ld_frag(Ks + (nw + 16 + ln) * 64 + kk * 32, khalf);
        s0 = WMMA_BF16(a, b0, s0);
        s1 = WMMA_BF16(a, b1, s1);
      }
      // probabilities -> Ps (bf16)
#pragma unroll
      for (int i = 0; i < 8; ++i) {
        const int m  = m0 + lm + i;
        const int gr = r0 + m;
        const float mh = Mst[h * 64 + m];
        const float li = Lst[h * 64 + m];
        const int c0 = j * 64 + nw + ln;
        const float p0 = (c0 <= gr) ? __expf(s0[i] * scale - mh) * li : 0.f;
        Ps[m * 64 + nw + ln] = f2bf(p0);
        const int c1 = c0 + 16;
        const float p1 = (c1 <= gr) ? __expf(s1[i] * scale - mh) * li : 0.f;
        Ps[m * 64 + nw + 16 + ln] = f2bf(p1);
      }
      __syncthreads();

      // amean stripe accumulation (exclusive ownership -> plain RMW)
      {
        const int row = tid & 63;
        const int cb  = (tid >> 6) * 16;
        float* gp = Amean + (size_t)(b * S_ + r0 + row) * S_ + j * 64 + cb;
        const unsigned short* pr = Ps + row * 64 + cb;
#pragma unroll
        for (int i = 0; i < 16; ++i) gp[i] += bf2f(pr[i]) * (1.0f / H_);
      }

      // o += P @ V  (A = P rows, B = Vt columns)
#pragma unroll
      for (int kk = 0; kk < 2; ++kk) {
        v16bf a  = ld_frag(Ps + (m0 + ln) * 64 + kk * 32, khalf);
        v16bf b0 = ld_frag(Vt + (nw + ln) * 64 + kk * 32, khalf);
        v16bf b1 = ld_frag(Vt + (nw + 16 + ln) * 64 + kk * 32, khalf);
        o0 = WMMA_BF16(a, b0, o0);
        o1 = WMMA_BF16(a, b1, o1);
      }
    }

    // write o tile: O[b, r0+m, h*64 + d]
#pragma unroll
    for (int i = 0; i < 8; ++i) {
      const int m = m0 + lm + i;
      float* op = O + (size_t)(b * S_ + r0 + m) * D_ + h * 64;
      op[nw + ln]      = o0[i];
      op[nw + 16 + ln] = o1[i];
    }
  }
}

// ---------------- host launcher ----------------

extern "C" void kernel_launch(void* const* d_in, const int* in_sizes, int n_in,
                              void* d_out, int out_size, void* d_ws, size_t ws_size,
                              hipStream_t stream) {
  (void)in_sizes; (void)n_in; (void)out_size; (void)ws_size;

  const float* x   = (const float*)d_in[0];
  const float* Wq  = (const float*)d_in[1];
  const float* bq  = (const float*)d_in[2];
  const float* Wkv = (const float*)d_in[3];
  const float* bkv = (const float*)d_in[4];

  float* O     = (float*)d_out;
  float* Amean = O + (size_t)B_ * S_ * D_;

  // workspace: q, k, v in bf16 [B,H,S,hd] layout (3 x 16 MB = 48 MB)
  const size_t qe = (size_t)B_ * H_ * S_ * HD_;
  unsigned short* qws = (unsigned short*)d_ws;
  unsigned short* kws = qws + qe;
  unsigned short* vws = kws + qe;

  // 1) zero the head-mean region (output buffer is poisoned)
  zero_f32<<<4096, 256, 0, stream>>>(Amean, (int)((size_t)B_ * S_ * S_ / 4));

  // 2) q = x @ Wq + bq
  qkv_gemm<<<dim3(D_ / 64, (B_ * S_) / 64), 256, 0, stream>>>(
      x, Wq, bq, qws, nullptr, D_, 0);

  // 3) kv = x @ Wkv + bkv -> k, v
  qkv_gemm<<<dim3((2 * D_) / 64, (B_ * S_) / 64), 256, 0, stream>>>(
      x, Wkv, bkv, kws, vws, 2 * D_, 1);

  // 4) causal attention + head-mean
  attn<<<dim3(S_ / 64, B_), 256, 0, stream>>>(qws, kws, vws, O, Amean);
}
